// GATLayer_77498389889093
// MI455X (gfx1250) — matrix operand order repaired and checked by reading, not compiled, not run on hardware
//
#include <hip/hip_runtime.h>
#include <hip/hip_fp16.h>

// ---------------------------------------------------------------------------
// GATv2 layer for MI455X (gfx1250, wave32, WMMA).
// N=50000 nodes, E=640000 edges, D=128, H=4 heads, C=32 channels/head.
//
// Projections x@W_l / x@W_r run as one fused WMMA GEMM over pre-packed f16
// fragments (single vector load per fragment). Edge softmax/aggregation uses
// atomic segment ops that ride the 192MB L2.
// ---------------------------------------------------------------------------

typedef __attribute__((ext_vector_type(16))) _Float16 v16h;
typedef __attribute__((ext_vector_type(8)))  float    v8f;

#define DIMS      128
#define HEADS     4
#define NEG_SLOPE 0.2f
#define LN_EPS    1e-5f

// ---- float atomic max via sign-split integer atomics (monotonic mapping) --
__device__ __forceinline__ void atomicMaxF(float* addr, float val) {
  if (val >= 0.0f) {
    atomicMax((int*)addr, __float_as_int(val));
  } else {
    atomicMin((unsigned int*)addr, __float_as_uint(val));
  }
}

// ---------------------------------------------------------------------------
// K0: zero/initialize accumulators. idx spans N*128 (covers all smaller bufs)
// ---------------------------------------------------------------------------
__global__ void __launch_bounds__(256)
k_init(float* __restrict__ out, float* __restrict__ attr_sum,
       float* __restrict__ cnt, float* __restrict__ amax,
       float* __restrict__ denom, int N_) {
  int i = blockIdx.x * 256 + threadIdx.x;
  if (i < N_ * DIMS) out[i] = 0.0f;
  if (i < N_) { attr_sum[i] = 0.0f; cnt[i] = 0.0f; }
  if (i < N_ * HEADS) { amax[i] = -1.0e30f; denom[i] = 0.0f; }
}

// ---------------------------------------------------------------------------
// K1: self-loop attr stats: attr_sum[d] += ea[e], cnt[d] += 1  (ED == 1)
// ---------------------------------------------------------------------------
__global__ void __launch_bounds__(256)
k_loop_stats(const int* __restrict__ edge_index, const float* __restrict__ edge_attr,
             float* __restrict__ attr_sum, float* __restrict__ cnt, int E_) {
  int e = blockIdx.x * 256 + threadIdx.x;
  if (e >= E_) return;
  int d = edge_index[E_ + e];             // dst row of (2,E)
  atomicAdd(&attr_sum[d], edge_attr[e]);
  atomicAdd(&cnt[d], 1.0f);
}

// ---------------------------------------------------------------------------
// Fragment layouts for v_wmma_f32_16x16x32_f16 (wave32):
//  A (16x32 MxK, 16-bit): lane l: M=l&15, kg=l>>4;
//    half j<8 :  K = kg*8 + j        (VGPRs 0..3)
//    half j>=8:  K = 16 + kg*8 + (j-8) (VGPRs 4..7)
//  B (32x16 KxN, 16-bit): lane l: N=l&15, kg=l>>4;
//    half pair j<4: K = kg*8 + 2j, 2j+1; j>=4: K = 16 + kg*8 + 2(j-4), +1
//  C/D (16x16 f32): lane l: N=l&15; VGPR i: M=(l>>4)*8 + i
// ---------------------------------------------------------------------------

// K2a: pack X (fp32, N x 128) into per-(mtile, kblock, lane) A fragments (f16)
__global__ void __launch_bounds__(256)
k_pack_a(const float* __restrict__ X, _Float16* __restrict__ Afrag, int nMt) {
  int idx = blockIdx.x * 256 + threadIdx.x;        // (mt, kb4, lane)
  if (idx >= nMt * 4 * 32) return;
  int lane = idx & 31;
  int kb4  = (idx >> 5) & 3;
  int mt   = idx >> 7;
  int m15 = lane & 15, kg = lane >> 4;
  const float* row = X + (size_t)(mt * 16 + m15) * DIMS + kb4 * 32;
  v16h f{};
  #pragma unroll
  for (int j = 0; j < 4; ++j) {
    float2 p0 = *(const float2*)(row + kg * 8 + 2 * j);
    f[2 * j]     = (_Float16)p0.x;
    f[2 * j + 1] = (_Float16)p0.y;
    float2 p1 = *(const float2*)(row + 16 + kg * 8 + 2 * j);
    f[8 + 2 * j]     = (_Float16)p1.x;
    f[8 + 2 * j + 1] = (_Float16)p1.y;
  }
  *(v16h*)(Afrag + (size_t)idx * 16) = f;
}

// K2b: pack [W_l | W_r] (each 128x128 fp32) into B fragments (f16, 64KB)
// mtile 0..7 -> W_l column tiles, 8..15 -> W_r column tiles.
__global__ void __launch_bounds__(256)
k_pack_b(const float* __restrict__ W_l, const float* __restrict__ W_r,
         _Float16* __restrict__ Bfrag) {
  int idx = blockIdx.x * 256 + threadIdx.x;        // (mtile, kb4, lane): 16*4*32
  if (idx >= 16 * 4 * 32) return;
  int lane  = idx & 31;
  int kb4   = (idx >> 5) & 3;
  int mtile = idx >> 7;
  const float* W = (mtile < 8) ? W_l : W_r;
  int ncol = (mtile & 7) * 16 + (lane & 15);
  int kg = lane >> 4;
  v16h f{};
  #pragma unroll
  for (int j = 0; j < 8; ++j) {
    int k = kb4 * 32 + ((j < 4) ? kg * 8 + 2 * j : 16 + kg * 8 + 2 * (j - 4));
    f[2 * j]     = (_Float16)W[(size_t)k * DIMS + ncol];
    f[2 * j + 1] = (_Float16)W[(size_t)(k + 1) * DIMS + ncol];
  }
  *(v16h*)(Bfrag + (size_t)idx * 16) = f;
}

// ---------------------------------------------------------------------------
// K2c: fused dual GEMM. Block = one 16-row stripe (mt). Wave w of 8 computes
// two 16x16 output tiles (waves 0-3 -> x_l cols, 4-7 -> x_r cols).
// Inner loop: 1 A-frag load + 2 B-frag loads + 2 WMMA.  8 v_wmma per wave.
// ---------------------------------------------------------------------------
__global__ void __launch_bounds__(256)
k_gemm_dual(const _Float16* __restrict__ Afrag, const _Float16* __restrict__ Bfrag,
            const float* __restrict__ b_l, const float* __restrict__ b_r,
            float* __restrict__ xl, float* __restrict__ xr, int nMt) {
  int w    = threadIdx.x >> 5;
  int lane = threadIdx.x & 31;
  int mt   = blockIdx.x;
  if (mt >= nMt) return;
  int mtile0 = 2 * w;
  int mtile1 = 2 * w + 1;

  v8f acc0 = {}, acc1 = {};
  const _Float16* abase = Afrag + ((size_t)mt * 4 * 32 + lane) * 16;
  #pragma unroll
  for (int kb4 = 0; kb4 < 4; ++kb4) {
    v16h a  = *(const v16h*)(abase + (size_t)kb4 * 32 * 16);
    v16h b0 = *(const v16h*)(Bfrag + (((size_t)mtile0 * 4 + kb4) * 32 + lane) * 16);
    v16h b1 = *(const v16h*)(Bfrag + (((size_t)mtile1 * 4 + kb4) * 32 + lane) * 16);
    acc0 = __builtin_amdgcn_wmma_f32_16x16x32_f16(false, a, false, b0,
                                                  (short)0, acc0, false, false);
    acc1 = __builtin_amdgcn_wmma_f32_16x16x32_f16(false, a, false, b1,
                                                  (short)0, acc1, false, false);
  }

  int m15 = lane & 15, kg = lane >> 4;
  float*       P    = (w < 4) ? xl  : xr;
  const float* bias = (w < 4) ? b_l : b_r;
  int col0 = (w & 3) * 32 + m15;
  int col1 = col0 + 16;
  float bb0 = bias[col0], bb1 = bias[col1];
  int mbase = mt * 16 + kg * 8;
  #pragma unroll
  for (int i = 0; i < 8; ++i) {
    P[(size_t)(mbase + i) * DIMS + col0] = acc0[i] + bb0;
    P[(size_t)(mbase + i) * DIMS + col1] = acc1[i] + bb1;
  }
}

// ---------------------------------------------------------------------------
// K3: per-edge attention logits. One wave per edge; lane handles 4 channels.
// alpha[e,h] = sum_c lrelu(xl[src]+xr[dst]+ea*We)[h,c] * att[h,c]
// Also atomic segment-max into amax[dst,h]. Self-loop attr = attr_sum/cnt.
// ---------------------------------------------------------------------------
__global__ void __launch_bounds__(256)
k_edge_alpha(const float* __restrict__ xl, const float* __restrict__ xr,
             const float* __restrict__ edge_attr, const float* __restrict__ attr_sum,
             const float* __restrict__ cnt, const int* __restrict__ edge_index,
             const float* __restrict__ W_e, const float* __restrict__ att,
             float* __restrict__ alpha, float* __restrict__ amax, int E_, int N_) {
  int wid  = (blockIdx.x * 256 + threadIdx.x) >> 5;
  int lane = threadIdx.x & 31;
  int Et = E_ + N_;
  if (wid >= Et) return;
  int s, d; float eav;
  if (wid < E_) { s = edge_index[wid]; d = edge_index[E_ + wid]; eav = edge_attr[wid]; }
  else          { s = wid - E_; d = s; eav = attr_sum[s] / fmaxf(cnt[s], 1.0f); }
  int cf = lane * 4;
  float4 vl = *(const float4*)(xl + (size_t)s * DIMS + cf);
  float4 vr = *(const float4*)(xr + (size_t)d * DIMS + cf);
  float4 we = *(const float4*)(W_e + cf);
  float4 at = *(const float4*)(att + cf);
  float m0 = vl.x + vr.x + eav * we.x;  m0 = m0 > 0.f ? m0 : m0 * NEG_SLOPE;
  float m1 = vl.y + vr.y + eav * we.y;  m1 = m1 > 0.f ? m1 : m1 * NEG_SLOPE;
  float m2 = vl.z + vr.z + eav * we.z;  m2 = m2 > 0.f ? m2 : m2 * NEG_SLOPE;
  float m3 = vl.w + vr.w + eav * we.w;  m3 = m3 > 0.f ? m3 : m3 * NEG_SLOPE;
  float p = m0 * at.x + m1 * at.y + m2 * at.z + m3 * at.w;
  // reduce 8 lanes (one head spans lanes h*8 .. h*8+7)
  p += __shfl_xor(p, 1, 32);
  p += __shfl_xor(p, 2, 32);
  p += __shfl_xor(p, 4, 32);
  if ((lane & 7) == 0) {
    int h = lane >> 3;
    alpha[(size_t)wid * HEADS + h] = p;
    atomicMaxF(&amax[(size_t)d * HEADS + h], p);
  }
}

// ---------------------------------------------------------------------------
// K4: ex = exp(alpha - amax[dst]); denom[dst,h] += ex.  One thread per (e,h).
// ---------------------------------------------------------------------------
__global__ void __launch_bounds__(256)
k_edge_exp(float* __restrict__ alpha, const float* __restrict__ amax,
           float* __restrict__ denom, const int* __restrict__ edge_index,
           int E_, int N_) {
  int idx = blockIdx.x * 256 + threadIdx.x;
  int Et = E_ + N_;
  if (idx >= Et * HEADS) return;
  int e = idx >> 2;          // HEADS == 4
  int h = idx & 3;
  int d = (e < E_) ? edge_index[E_ + e] : (e - E_);
  float ex = __expf(alpha[idx] - amax[(size_t)d * HEADS + h]);
  alpha[idx] = ex;
  atomicAdd(&denom[(size_t)d * HEADS + h], ex);
}

// ---------------------------------------------------------------------------
// K5: out[dst] += xl[src] * (ex / denom[dst]).  One wave per edge.
// ---------------------------------------------------------------------------
__global__ void __launch_bounds__(256)
k_edge_agg(const float* __restrict__ xl, const float* __restrict__ alpha,
           const float* __restrict__ denom, const int* __restrict__ edge_index,
           float* __restrict__ out, int E_, int N_) {
  int wid  = (blockIdx.x * 256 + threadIdx.x) >> 5;
  int lane = threadIdx.x & 31;
  int Et = E_ + N_;
  if (wid >= Et) return;
  int s, d;
  if (wid < E_) { s = edge_index[wid]; d = edge_index[E_ + wid]; }
  else          { s = wid - E_;        d = s; }
  int cf = lane * 4;
  int h  = lane >> 3;
  float a = alpha[(size_t)wid * HEADS + h] / denom[(size_t)d * HEADS + h];
  float4 vl = *(const float4*)(xl + (size_t)s * DIMS + cf);
  float* ob = out + (size_t)d * DIMS + cf;
  atomicAdd(ob + 0, vl.x * a);
  atomicAdd(ob + 1, vl.y * a);
  atomicAdd(ob + 2, vl.z * a);
  atomicAdd(ob + 3, vl.w * a);
}

// ---------------------------------------------------------------------------
// K6: out = LayerNorm(elu(out + bias) + x) * gamma + beta.  One wave per node.
// ---------------------------------------------------------------------------
__global__ void __launch_bounds__(256)
k_finalize(float* __restrict__ out, const float* __restrict__ x,
           const float* __restrict__ bias, const float* __restrict__ gamma,
           const float* __restrict__ beta, int N_) {
  int node = (blockIdx.x * 256 + threadIdx.x) >> 5;
  int lane = threadIdx.x & 31;
  if (node >= N_) return;
  int cf = lane * 4;
  float4 v = *(const float4*)(out + (size_t)node * DIMS + cf);
  float4 b = *(const float4*)(bias + cf);
  float4 xx = *(const float4*)(x + (size_t)node * DIMS + cf);
  float t;
  t = v.x + b.x; v.x = (t > 0.f ? t : __expf(t) - 1.0f) + xx.x;
  t = v.y + b.y; v.y = (t > 0.f ? t : __expf(t) - 1.0f) + xx.y;
  t = v.z + b.z; v.z = (t > 0.f ? t : __expf(t) - 1.0f) + xx.z;
  t = v.w + b.w; v.w = (t > 0.f ? t : __expf(t) - 1.0f) + xx.w;
  float s1 = v.x + v.y + v.z + v.w;
  float s2 = v.x * v.x + v.y * v.y + v.z * v.z + v.w * v.w;
  #pragma unroll
  for (int off = 1; off < 32; off <<= 1) {
    s1 += __shfl_xor(s1, off, 32);
    s2 += __shfl_xor(s2, off, 32);
  }
  float mu  = s1 * (1.0f / DIMS);
  float var = s2 * (1.0f / DIMS) - mu * mu;
  float inv = rsqrtf(var + LN_EPS);
  float4 g  = *(const float4*)(gamma + cf);
  float4 be = *(const float4*)(beta + cf);
  v.x = (v.x - mu) * inv * g.x + be.x;
  v.y = (v.y - mu) * inv * g.y + be.y;
  v.z = (v.z - mu) * inv * g.z + be.z;
  v.w = (v.w - mu) * inv * g.w + be.w;
  *(float4*)(out + (size_t)node * DIMS + cf) = v;
}

// ---------------------------------------------------------------------------
extern "C" void kernel_launch(void* const* d_in, const int* in_sizes, int n_in,
                              void* d_out, int out_size, void* d_ws, size_t ws_size,
                              hipStream_t stream) {
  const float* x         = (const float*)d_in[0];
  const int*   edge_index= (const int*)  d_in[1];
  const float* edge_attr = (const float*)d_in[2];
  const float* W_l       = (const float*)d_in[3];
  const float* b_l       = (const float*)d_in[4];
  const float* W_r       = (const float*)d_in[5];
  const float* b_r       = (const float*)d_in[6];
  const float* W_e       = (const float*)d_in[7];
  const float* att       = (const float*)d_in[8];
  const float* bias      = (const float*)d_in[9];
  const float* gamma     = (const float*)d_in[10];
  const float* beta      = (const float*)d_in[11];
  float* out = (float*)d_out;

  const int N_  = in_sizes[0] / DIMS;   // 50000
  const int E_  = in_sizes[1] / 2;      // 640000
  const int Et  = E_ + N_;
  const int nMt = N_ / 16;              // 3125 row stripes (N is a multiple of 16)

  // Workspace carve-up (256B-aligned chunks)
  char* ws = (char*)d_ws;
  size_t off = 0;
  auto carve = [&](size_t bytes) -> void* {
    void* p = (void*)(ws + off);
    off += (bytes + 255) & ~(size_t)255;
    return p;
  };
  float*     xl       = (float*)carve((size_t)N_ * DIMS * 4);
  float*     xr       = (float*)carve((size_t)N_ * DIMS * 4);
  float*     alpha    = (float*)carve((size_t)Et * HEADS * 4);
  float*     amax     = (float*)carve((size_t)N_ * HEADS * 4);
  float*     denom    = (float*)carve((size_t)N_ * HEADS * 4);
  float*     attr_sum = (float*)carve((size_t)N_ * 4);
  float*     cnt      = (float*)carve((size_t)N_ * 4);
  _Float16*  Afrag    = (_Float16*)carve((size_t)nMt * 4 * 32 * 16 * 2); // 12.8MB
  _Float16*  Bfrag    = (_Float16*)carve((size_t)16 * 4 * 32 * 16 * 2);  // 64KB
  (void)ws_size;

  // K0: init accumulators + zero output
  k_init<<<(N_ * DIMS + 255) / 256, 256, 0, stream>>>(out, attr_sum, cnt, amax, denom, N_);
  // K1: self-loop attr sums/counts
  k_loop_stats<<<(E_ + 255) / 256, 256, 0, stream>>>(edge_index, edge_attr, attr_sum, cnt, E_);
  // K2: pack fragments, then fused dual WMMA GEMM
  k_pack_a<<<(nMt * 4 * 32 + 255) / 256, 256, 0, stream>>>(x, Afrag, nMt);
  k_pack_b<<<(16 * 4 * 32 + 255) / 256, 256, 0, stream>>>(W_l, W_r, Bfrag);
  k_gemm_dual<<<nMt, 256, 0, stream>>>(Afrag, Bfrag, b_l, b_r, xl, xr, nMt);
  // K3: attention logits + segment max
  k_edge_alpha<<<(Et + 7) / 8, 256, 0, stream>>>(xl, xr, edge_attr, attr_sum, cnt,
                                                 edge_index, W_e, att, alpha, amax, E_, N_);
  // K4: exp + segment sum
  k_edge_exp<<<(Et * HEADS + 255) / 256, 256, 0, stream>>>(alpha, amax, denom,
                                                           edge_index, E_, N_);
  // K5: weighted scatter-aggregate
  k_edge_agg<<<(Et + 7) / 8, 256, 0, stream>>>(xl, alpha, denom, edge_index, out, E_, N_);
  // K6: bias + ELU + residual + LayerNorm
  k_finalize<<<(N_ + 7) / 8, 256, 0, stream>>>(out, x, bias, gamma, beta, N_);
}
